// Basic097_6408091206334
// MI455X (gfx1250) — matrix operand order repaired and checked by reading, not compiled, not run on hardware
//
#include <hip/hip_runtime.h>
#include <hip/hip_bf16.h>
#include <math.h>

// Problem dims (must match the reference)
#define VOCAB  10000
#define HID    1024
#define OUTN   10000
#define BATCH  32
#define SEQ    256

// out_gemm tiling
#define NPANEL 80            // N columns per block (5 tiles of 16); 10000/80 = 125
#define MPANEL 128           // M rows per block (8 waves x 16)

typedef __attribute__((ext_vector_type(16))) __bf16   v16bf;
typedef __attribute__((ext_vector_type(8)))  float    v8f;
typedef __attribute__((ext_vector_type(8)))  int      v8i;
typedef __attribute__((ext_vector_type(4)))  unsigned v4u;
typedef __attribute__((ext_vector_type(8)))  int      v8s;
typedef __attribute__((ext_vector_type(4)))  int      v4s;

// ---------- helpers ----------

static __device__ __forceinline__ __bf16 f2bf(float f) {
    unsigned u = __builtin_bit_cast(unsigned, f);
    u = (u + 0x7FFFu + ((u >> 16) & 1u)) >> 16;   // round-to-nearest-even
    unsigned short s = (unsigned short)u;
    return __builtin_bit_cast(__bf16, s);
}

// A fragment: 16x32 bf16, row-major source (row = one M line, K contiguous).
// ISA layout: lanes 0-15 (hi=0): V0..3 hold K=0..7 pairs, V4..7 hold K=16..23;
//             lanes 16-31 (hi=1): +8.
static __device__ __forceinline__ v8i load_a_frag(const __bf16* __restrict__ row, int hi) {
    const __bf16* p = row + 8 * hi;
    int4 x0 = *(const int4*)(p);        // K = 8h .. 8h+7
    int4 x1 = *(const int4*)(p + 16);   // K = 16+8h .. 16+8h+7
    v8i r = { x0.x, x0.y, x0.z, x0.w, x1.x, x1.y, x1.z, x1.w };
    return r;
}

// B fragment: 32x16 bf16 from a K-major (transposed) weight column.
// ISA layout: lanes 0-15 (hi=0): V0..3 hold K=0..7, V4..7 hold K=16..23; lanes 16-31: +8.
static __device__ __forceinline__ v8i load_b_frag(const __bf16* __restrict__ col, int hi) {
    const __bf16* p = col + 8 * hi;
    int4 x0 = *(const int4*)(p);        // K = 8h .. 8h+7
    int4 x1 = *(const int4*)(p + 16);   // K = 16+8h .. 16+8h+7
    v8i r = { x0.x, x0.y, x0.z, x0.w, x1.x, x1.y, x1.z, x1.w };
    return r;
}

static __device__ __forceinline__ v8f wmma_bf16(v8i a, v8i b, v8f c) {
    return __builtin_amdgcn_wmma_f32_16x16x32_bf16(
        /*neg_a=*/false, __builtin_bit_cast(v16bf, a),
        /*neg_b=*/false, __builtin_bit_cast(v16bf, b),
        /*c_mod=*/(short)0, c, /*reuse_a=*/false, /*reuse_b=*/false);
}

// ---------- prep kernels ----------

// WT[n*K + k] = bf16(W[k*N + n])  (produce K-major columns for B frags)
__global__ void transpose_to_bf16(const float* __restrict__ W, __bf16* __restrict__ WT,
                                  int K, int N) {
    int n = blockIdx.x * blockDim.x + threadIdx.x;
    int k = blockIdx.y * blockDim.y + threadIdx.y;
    if (n < N && k < K) WT[(size_t)n * K + k] = f2bf(W[(size_t)k * N + n]);
}

__global__ void state_to_bf16(const float* __restrict__ s, __bf16* __restrict__ h0) {
    int i = blockIdx.x * blockDim.x + threadIdx.x;     // BATCH*HID threads
    h0[i] = f2bf(s[i]);
}

// ---------- recurrence: H_next = tanh(W_xh[x_t] + H_prev @ W_hh + b_h) ----------
// grid.x = HID/16; block = 256 threads = 8 waves.
// wave = mt (M tile 0/1) + 2*kw (K split 0..3); each wave does 8 dependent WMMAs,
// partials reduced through LDS -> short serial chain per timestep.
__global__ __launch_bounds__(256) void rnn_step(
    const int*   __restrict__ x,        // (BATCH, SEQ) int32
    int t,
    const float* __restrict__ W_xh,     // (VOCAB, HID) f32
    const float* __restrict__ b_h,      // (HID,) f32
    const __bf16* __restrict__ W_hhT,   // (HID n-major, HID k) bf16
    const __bf16* __restrict__ H_prev,  // (BATCH, HID) bf16
    __bf16* __restrict__ H_next,        // (BATCH, HID) bf16
    float*  __restrict__ H_f32)         // (BATCH, HID) f32 (H_final slot; last write wins)
{
    __shared__ float red[8 * 256];      // [wave][lane*8 + r]

    const int lane = threadIdx.x & 31;
    const int wave = threadIdx.x >> 5;
    const int mt   = wave & 1;          // M tile (batch rows 0-15 / 16-31)
    const int kw   = wave >> 1;         // K split 0..3 (256 each)
    const int lo   = lane & 15;
    const int hi   = lane >> 4;
    const int n0   = blockIdx.x * 16;

    const __bf16* arow = H_prev + (size_t)(mt * 16 + lo) * HID;
    const __bf16* bcol = W_hhT + (size_t)(n0 + lo) * HID;

    v8f acc = {0.f, 0.f, 0.f, 0.f, 0.f, 0.f, 0.f, 0.f};
    const int kbeg = kw * 256;
#pragma unroll
    for (int kk = 0; kk < 256; kk += 32) {
        v8i a = load_a_frag(arow + kbeg + kk, hi);
        v8i b = load_b_frag(bcol + kbeg + kk, hi);
        acc = wmma_bf16(a, b, acc);
    }

    float* myred = &red[wave * 256 + lane * 8];
    float4 s0; s0.x = acc[0]; s0.y = acc[1]; s0.z = acc[2]; s0.w = acc[3];
    float4 s1; s1.x = acc[4]; s1.y = acc[5]; s1.z = acc[6]; s1.w = acc[7];
    *(float4*)(myred)     = s0;
    *(float4*)(myred + 4) = s1;
    __syncthreads();

    if (kw == 0) {
        float sum[8];
#pragma unroll
        for (int r = 0; r < 8; ++r) sum[r] = 0.f;
#pragma unroll
        for (int j = 0; j < 4; ++j) {
            const float* pr = &red[(mt + 2 * j) * 256 + lane * 8];
            float4 u0 = *(const float4*)(pr);
            float4 u1 = *(const float4*)(pr + 4);
            sum[0] += u0.x; sum[1] += u0.y; sum[2] += u0.z; sum[3] += u0.w;
            sum[4] += u1.x; sum[5] += u1.y; sum[6] += u1.z; sum[7] += u1.w;
        }
        const int nn = n0 + lo;
        const float bh = b_h[nn];
#pragma unroll
        for (int r = 0; r < 8; ++r) {
            const int mm = mt * 16 + r + 8 * hi;            // batch row
            const int id = x[mm * SEQ + t];                 // token id
            const float v = sum[r] + W_xh[(size_t)id * HID + nn] + bh;
            const float h = tanhf(v);
            H_next[(size_t)mm * HID + nn] = f2bf(h);
            H_f32 [(size_t)mm * HID + nn] = h;
        }
    }
}

// ---------- output GEMM: Y = H_all @ W_hq + b_q  (8192 x 10000, K=1024) ----------
// grid = (OUTN/NPANEL = 125, 8192/MPANEL = 64); block = 256 threads = 8 waves.
// The block's B panel (NPANEL rows of W_hqT, K-major, 160 KB) is staged into LDS
// with one Tensor Data Mover descriptor (wave 0 issues; TENSORcnt + barrier).
// Each wave owns a 16(M) x 80(N) strip: one A frag feeds 5 WMMAs per K step.
__global__ __launch_bounds__(256) void out_gemm(
    const __bf16* __restrict__ Hseq,    // (SEQ*BATCH, HID) bf16 (slot-1 base)
    const __bf16* __restrict__ W_hqT,   // (OUTN n-major, HID k) bf16
    const float*  __restrict__ b_q,     // (OUTN,) f32
    float* __restrict__ Y)              // (SEQ*BATCH, OUTN) f32
{
    __shared__ __bf16 Bpanel[NPANEL * HID];   // 160 KB (of 320 KB WGP LDS)

    const int lane = threadIdx.x & 31;
    const int wave = threadIdx.x >> 5;
    const int lo   = lane & 15;
    const int hi   = lane >> 4;
    const int n0   = blockIdx.x * NPANEL;
    const int m0   = blockIdx.y * MPANEL + wave * 16;

    const __bf16* gB = W_hqT + (size_t)n0 * HID;   // B panel: NPANEL x HID, K-major

#if __has_builtin(__builtin_amdgcn_tensor_load_to_lds) && __has_builtin(__builtin_amdgcn_s_wait_tensorcnt)
    if (wave == 0) {
        // ---- Tensor DMA descriptor (CDNA5 ISA) ----
        unsigned lds_addr = (unsigned)(size_t)(&Bpanel[0]);        // LDS byte offset
        unsigned long long ga = (unsigned long long)(size_t)gB;    // 57-bit global addr
        // Group 0: [count=1 | lds_addr | global_addr | type=2]
        v4u g0 = { 1u,
                   lds_addr,
                   (unsigned)(ga & 0xFFFFFFFFu),
                   (unsigned)((ga >> 32) & 0x01FFFFFFu) | (2u << 30) };
        // Group 1: data_size=1 (2B); tensor = 1024 x NPANEL; tile = 1024 x NPANEL;
        //          tensor_dim0_stride = 1024 elements; 2D (tile_dim2 = 0).
        v8s g1 = { (int)(1u << 16),                       // wg_mask=0, data_size=2B
                   (int)(1024u << 16),                    // tensor_dim0[15:0]<<16
                   (int)((unsigned)NPANEL << 16),         // dim0[31:16]=0 | tensor_dim1[15:0]
                   (int)(1024u << 16),                    // dim1[31:16]=0 | tile_dim0=1024
                   (int)NPANEL,                           // tile_dim1 | tile_dim2=0
                   (int)1024,                             // tensor_dim0_stride[31:0]
                   0,                                     // stride[47:32] | dim1_stride lo
                   0 };
        v4s gz4 = { 0, 0, 0, 0 };                         // groups 2/3 unused (2D tile)
        v8s gz8 = { 0, 0, 0, 0, 0, 0, 0, 0 };             // extra descriptor words unused
        __builtin_amdgcn_tensor_load_to_lds(g0, g1, gz4, gz4, gz8, 0);
        __builtin_amdgcn_s_wait_tensorcnt((short)0);
    }
#else
    // Fallback: cooperative vector copy global -> LDS
    {
        const int4* src = (const int4*)gB;
        int4* dst = (int4*)&Bpanel[0];
        for (int i = threadIdx.x; i < NPANEL * HID / 8; i += 256) dst[i] = src[i];
    }
#endif
    __syncthreads();

    const __bf16* arow = Hseq + (size_t)(m0 + lo) * HID;

    v8f acc0 = {0.f,0.f,0.f,0.f,0.f,0.f,0.f,0.f};
    v8f acc1 = acc0, acc2 = acc0, acc3 = acc0, acc4 = acc0;

#pragma unroll 2
    for (int kb = 0; kb < HID; kb += 32) {
        v8i a = load_a_frag(arow + kb, hi);
        v8i b0 = load_b_frag(&Bpanel[(0 * 16 + lo) * HID + kb], hi);
        v8i b1 = load_b_frag(&Bpanel[(1 * 16 + lo) * HID + kb], hi);
        v8i b2 = load_b_frag(&Bpanel[(2 * 16 + lo) * HID + kb], hi);
        v8i b3 = load_b_frag(&Bpanel[(3 * 16 + lo) * HID + kb], hi);
        v8i b4 = load_b_frag(&Bpanel[(4 * 16 + lo) * HID + kb], hi);
        acc0 = wmma_bf16(a, b0, acc0);
        acc1 = wmma_bf16(a, b1, acc1);
        acc2 = wmma_bf16(a, b2, acc2);
        acc3 = wmma_bf16(a, b3, acc3);
        acc4 = wmma_bf16(a, b4, acc4);
    }

#pragma unroll
    for (int j = 0; j < 5; ++j) {
        v8f acc = (j == 0) ? acc0 : (j == 1) ? acc1 : (j == 2) ? acc2 : (j == 3) ? acc3 : acc4;
        const int nn = n0 + j * 16 + lo;
        const float bq = b_q[nn];
#pragma unroll
        for (int r = 0; r < 8; ++r) {
            const int mm = m0 + r + 8 * hi;
            Y[(size_t)mm * OUTN + nn] = acc[r] + bq;
        }
    }
}

// ---------- launcher ----------

extern "C" void kernel_launch(void* const* d_in, const int* in_sizes, int n_in,
                              void* d_out, int out_size, void* d_ws, size_t ws_size,
                              hipStream_t stream) {
    const int*   x     = (const int*)  d_in[0];   // (BATCH, SEQ)
    const float* state = (const float*)d_in[1];   // (BATCH, HID)
    const float* W_xh  = (const float*)d_in[2];   // (VOCAB, HID)
    const float* W_hh  = (const float*)d_in[3];   // (HID, HID)
    const float* b_h   = (const float*)d_in[4];   // (HID,)
    const float* W_hq  = (const float*)d_in[5];   // (HID, OUTN)
    const float* b_q   = (const float*)d_in[6];   // (OUTN,)

    float* Y       = (float*)d_out;                         // (SEQ*BATCH, OUTN)
    float* H_final = Y + (size_t)SEQ * BATCH * OUTN;        // (BATCH, HID)

    // Workspace layout (bf16): H history (SEQ+1 slots), W_hh^T, W_hq^T
    char* ws = (char*)d_ws;
    __bf16* H_all = (__bf16*)ws;
    size_t offWhh = ((size_t)(SEQ + 1) * BATCH * HID * 2 + 255) & ~(size_t)255;
    __bf16* W_hhT = (__bf16*)(ws + offWhh);
    size_t offWhq = offWhh + (size_t)HID * HID * 2;
    __bf16* W_hqT = (__bf16*)(ws + offWhq);
    (void)ws_size; (void)in_sizes; (void)n_in; (void)out_size;

    // 1) prep: bf16 state + transposed bf16 weights
    state_to_bf16<<<(BATCH * HID) / 256, 256, 0, stream>>>(state, H_all);
    {
        dim3 b(32, 8), g((HID + 31) / 32, (HID + 7) / 8);
        transpose_to_bf16<<<g, b, 0, stream>>>(W_hh, W_hhT, HID, HID);
    }
    {
        dim3 b(32, 8), g((OUTN + 31) / 32, (HID + 7) / 8);
        transpose_to_bf16<<<g, b, 0, stream>>>(W_hq, W_hqT, HID, OUTN);
    }

    // 2) serial recurrence: 256 dependent steps
    for (int t = 0; t < SEQ; ++t) {
        const __bf16* Hp = H_all + (size_t)t       * BATCH * HID;
        __bf16*       Hn = H_all + (size_t)(t + 1) * BATCH * HID;
        rnn_step<<<HID / 16, 256, 0, stream>>>(x, t, W_xh, b_h, W_hhT, Hp, Hn, H_final);
    }

    // 3) one big output GEMM over all timesteps (TDM-staged B panel)
    out_gemm<<<dim3(OUTN / NPANEL, (SEQ * BATCH) / MPANEL), 256, 0, stream>>>(
        H_all + (size_t)BATCH * HID, W_hqT, b_q, Y);
}
